// LSTM_24232205484458
// MI455X (gfx1250) — matrix operand order repaired and checked by reading, not compiled, not run on hardware
//
#include <hip/hip_runtime.h>
#include <hip/hip_bf16.h>

// ---------------------------------------------------------------------------
// MI455X (gfx1250) LSTM:
//  - k_cvt / k_pack : fp32 -> bf16, weights pre-swizzled into WMMA B-fragment
//                     order (tiles of 32 lanes x 16 halves, contiguous 1KB).
//  - k_xg           : big parallel WMMA GEMM  xg = x @ [Wix|Wfx|Wcx|Wox].
//  - k_lstm         : persistent kernel, 56 WGs, each stages 256KB of packed
//                     recurrent weights into its 320KB LDS ONCE, then runs the
//                     512-step recurrence with a global epoch barrier between
//                     the 3 dependent phases per step.
// All matrix math lowers to v_wmma_f32_16x16x32_bf16 (wave32).
// ---------------------------------------------------------------------------

typedef __attribute__((ext_vector_type(16))) __bf16        v16bf;
typedef __attribute__((ext_vector_type(8)))  float         v8f;
typedef __attribute__((ext_vector_type(4)))  unsigned int  u32x4;

union Frag { u32x4 u[2]; v16bf v; };

__device__ __forceinline__ v8f wmma_bf16(v16bf a, v16bf b, v8f c) {
  return __builtin_amdgcn_wmma_f32_16x16x32_bf16(false, a, false, b,
                                                 (short)0, c, false, false);
}

static constexpr int NWG = 56;   // persistent grid: 16 i + 16 f + 8 cand + 8 oh + 8 oc

// ---------------- init: zero h, c, barrier counter ----------------
__global__ void k_init(__bf16* hbf, __bf16* cbf, float* cf, unsigned* ctr) {
  int i = blockIdx.x * 256 + threadIdx.x;          // 128*256 = 32768 = B*H
  if (i < 32768) { hbf[i] = (__bf16)0.f; cbf[i] = (__bf16)0.f; cf[i] = 0.f; }
  if (i == 0) *ctr = 0u;
}

// ---------------- fp32 -> bf16 convert (x) ----------------
__global__ void k_cvt(const float* __restrict__ s, __bf16* __restrict__ d, int n) {
  int i = (blockIdx.x * 256 + threadIdx.x) * 4;
  if (i + 3 < n) {
    d[i + 0] = (__bf16)s[i + 0];
    d[i + 1] = (__bf16)s[i + 1];
    d[i + 2] = (__bf16)s[i + 2];
    d[i + 3] = (__bf16)s[i + 3];
  }
}

// ---------------- pack one [1024,1024] fp32 weight into bf16 B-fragment order
// B operand (KxN 32x16 tile, 16-bit): lane = (n%16) | (((k%32>>3)&1)<<4),
// half-index j = (k%8) | ((k%32>=16)<<3).  Tile = 32 lanes * 16 halves = 512 elems = 1KB.
// Tiles stored nt-major: index ((nt*32 + kt)*32 + lane)*16 + j.
__global__ void k_pack(const float* __restrict__ src, __bf16* __restrict__ dst) {
  int e = blockIdx.x * 256 + threadIdx.x;          // 4096*256 = 1M elements
  int k = e >> 10, n = e & 1023;
  int kt = k >> 5, kk = k & 31, nt = n >> 4;
  int lane = (n & 15) | (((kk >> 3) & 1) << 4);
  int j = (kk & 7) | ((kk >> 4) << 3);
  dst[(((size_t)nt * 32 + kt) * 32 + lane) * 16 + j] = (__bf16)src[e];
}

// ---------------- xg = x @ Wx  :  [16384 x 1024] * [1024 x 4096] ----------------
// One 16x16 output tile per wave; A from x_bf16 (A-fragment pattern), B from
// packed weights (contiguous 1KB tiles).  Output layout: xg[t][gate][b][H] bf16.
__global__ void k_xg(const __bf16* __restrict__ xbf,
                     const __bf16* __restrict__ pwx,   // 4 packed gate matrices
                     __bf16* __restrict__ xg) {
  const int wave = threadIdx.x >> 5, lane = threadIdx.x & 31;
  const int tile = blockIdx.x * 8 + wave;          // < 1024 * 256
  const int rt = tile >> 8, ct = tile & 255;
  const int g = ct >> 6, nt = ct & 63;
  const __bf16* pw = pwx + (size_t)g * 1048576 + (size_t)nt * 16384;
  const __bf16* arow =
      xbf + (size_t)(rt * 16 + (lane & 15)) * 1024 + ((lane >> 4) << 3);
  v8f acc = {};
  for (int kt = 0; kt < 32; ++kt) {
    Frag a, b;
    a.u[0] = *reinterpret_cast<const u32x4*>(arow + kt * 32);
    a.u[1] = *reinterpret_cast<const u32x4*>(arow + kt * 32 + 16);
    const __bf16* bp = pw + kt * 512 + lane * 16;
    b.u[0] = *reinterpret_cast<const u32x4*>(bp);
    b.u[1] = *reinterpret_cast<const u32x4*>(bp + 8);
    acc = wmma_bf16(a.v, b.v, acc);
  }
  const int t = rt >> 1, b0 = (rt & 1) << 4;
  const int col = (nt << 4) + (lane & 15);
  const size_t base = (size_t)(t * 4 + g) * 32;
  for (int v = 0; v < 8; ++v) {
    int row = b0 + v + ((lane >> 4) << 3);
    xg[(base + row) * 1024 + col] = (__bf16)acc[v];
  }
}

// ---------------- global epoch barrier ----------------
__device__ __forceinline__ void gbar(unsigned* ctr, unsigned target) {
  __threadfence();                 // release my writes to agent scope
  __syncthreads();
  if (threadIdx.x == 0) {
    __hip_atomic_fetch_add(ctr, 1u, __ATOMIC_RELEASE, __HIP_MEMORY_SCOPE_AGENT);
    while (__hip_atomic_load(ctr, __ATOMIC_ACQUIRE, __HIP_MEMORY_SCOPE_AGENT) < target)
      __builtin_amdgcn_s_sleep(2);
  }
  __syncthreads();
  __threadfence();                 // acquire: invalidate L0 so we see peers' data
}

// ---------------- persistent recurrent kernel ----------------
// WG 0-15 : i gate  (K = h|c = 2048, 64 cols each, weights W_ih|W_ic in LDS)
// WG 16-31: f gate  (same shape, W_fh|W_fc)
// WG 32-39: cand    (K = 1024, 128 cols each, W_ch)
// WG 40-47: h@W_oh  (K = 1024, 128 cols each, W_oh)
// WG 48-55: phase2 c_new@W_oc + o/h epilogue (W_oc)
__global__ void __launch_bounds__(256, 1)
k_lstm(const __bf16* __restrict__ xg,
       const __bf16* __restrict__ pw_ih, const __bf16* __restrict__ pw_ic,
       const __bf16* __restrict__ pw_fh, const __bf16* __restrict__ pw_fc,
       const __bf16* __restrict__ pw_ch, const __bf16* __restrict__ pw_oh,
       const __bf16* __restrict__ pw_oc,
       const float* __restrict__ b_i, const float* __restrict__ b_f,
       const float* __restrict__ b_c, const float* __restrict__ b_o,
       __bf16* hbf, __bf16* cbf, float* cf,
       float* Gi, float* Gf, float* Gc, float* Goh,
       unsigned* ctr, float* out) {
  extern __shared__ __bf16 lds[];                  // 256 KB packed weights
  const int tid = threadIdx.x, wave = tid >> 5, lane = tid & 31;
  const int wg = blockIdx.x;
  const int koff = (lane >> 4) << 3;               // A-fragment k sub-offset
  const int clo = lane & 15;

  // ---- stage this WG's weights into LDS once (packed source is contiguous) ----
  if (wg < 32) {
    const int w = wg & 15;
    const __bf16* ph = (wg < 16) ? pw_ih : pw_fh;
    const __bf16* pc = (wg < 16) ? pw_ic : pw_fc;
    for (int l = 0; l < 4; ++l) {
      const u32x4* s0 = reinterpret_cast<const u32x4*>(ph + (size_t)(4 * w + l) * 16384);
      const u32x4* s1 = reinterpret_cast<const u32x4*>(pc + (size_t)(4 * w + l) * 16384);
      for (int i = tid; i < 2048; i += 256) {
        *reinterpret_cast<u32x4*>(lds + (size_t)l * 32768 + (size_t)i * 8) = s0[i];
        *reinterpret_cast<u32x4*>(lds + (size_t)l * 32768 + 16384 + (size_t)i * 8) = s1[i];
      }
    }
  } else {
    const __bf16* pw = (wg < 40) ? pw_ch : (wg < 48 ? pw_oh : pw_oc);
    const int w = (wg - 32) & 7;
    for (int l = 0; l < 8; ++l) {
      const u32x4* s0 = reinterpret_cast<const u32x4*>(pw + (size_t)(8 * w + l) * 16384);
      for (int i = tid; i < 2048; i += 256)
        *reinterpret_cast<u32x4*>(lds + (size_t)l * 16384 + (size_t)i * 8) = s0[i];
    }
  }
  __syncthreads();

  unsigned ep = 0;
  for (int t = 0; t < 512; ++t) {
    // ================= phase 1: i, f, cand, h@W_oh =================
    if (wg < 32) {                                 // i / f gate: 1 tile per wave
      const int gate = wg >> 4;                    // 0=i, 1=f
      const int w = wg & 15;
      const int ntl = wave >> 1, mtile = wave & 1;
      const int col = w * 64 + ntl * 16 + clo;
      const size_t rowoff = (size_t)(mtile * 16 + clo) * 1024;
      v8f acc = {};
      for (int kt = 0; kt < 32; ++kt) {            // h part
        Frag a, b;
        const int kb = (kt << 5) + koff;
        a.u[0] = *reinterpret_cast<const u32x4*>(hbf + rowoff + kb);
        a.u[1] = *reinterpret_cast<const u32x4*>(hbf + rowoff + kb + 16);
        const __bf16* bp = lds + (size_t)(ntl * 64 + kt) * 512 + lane * 16;
        b.u[0] = *reinterpret_cast<const u32x4*>(bp);
        b.u[1] = *reinterpret_cast<const u32x4*>(bp + 8);
        acc = wmma_bf16(a.v, b.v, acc);
      }
      for (int kt = 0; kt < 32; ++kt) {            // peephole c part
        Frag a, b;
        const int kb = (kt << 5) + koff;
        a.u[0] = *reinterpret_cast<const u32x4*>(cbf + rowoff + kb);
        a.u[1] = *reinterpret_cast<const u32x4*>(cbf + rowoff + kb + 16);
        const __bf16* bp = lds + (size_t)(ntl * 64 + 32 + kt) * 512 + lane * 16;
        b.u[0] = *reinterpret_cast<const u32x4*>(bp);
        b.u[1] = *reinterpret_cast<const u32x4*>(bp + 8);
        acc = wmma_bf16(a.v, b.v, acc);
      }
      const size_t xb = (size_t)(t * 4 + gate) * 32;
      for (int v = 0; v < 8; ++v) {
        int row = mtile * 16 + v + koff;
        float pre = acc[v] + (float)xg[(xb + row) * 1024 + col];
        if (gate == 0) {
          pre += b_i[col];
          Gi[row * 1024 + col] = 1.f / (1.f + __expf(-pre));
        } else {
          pre += b_f[row * 1024 + col];            // per-batch forget bias
          Gf[row * 1024 + col] = 1.f / (1.f + __expf(-pre));
        }
      }
    } else if (wg < 48) {                          // cand / oh: 2 tiles per wave
      const int isOh = (wg >= 40);
      const int w = (wg - 32) & 7;
      const int col = w * 128 + wave * 16 + clo;
      for (int mtile = 0; mtile < 2; ++mtile) {
        const size_t rowoff = (size_t)(mtile * 16 + clo) * 1024;
        v8f acc = {};
        for (int kt = 0; kt < 32; ++kt) {
          Frag a, b;
          const int kb = (kt << 5) + koff;
          a.u[0] = *reinterpret_cast<const u32x4*>(hbf + rowoff + kb);
          a.u[1] = *reinterpret_cast<const u32x4*>(hbf + rowoff + kb + 16);
          const __bf16* bp = lds + (size_t)(wave * 32 + kt) * 512 + lane * 16;
          b.u[0] = *reinterpret_cast<const u32x4*>(bp);
          b.u[1] = *reinterpret_cast<const u32x4*>(bp + 8);
          acc = wmma_bf16(a.v, b.v, acc);
        }
        const size_t xb = (size_t)(t * 4 + (isOh ? 3 : 2)) * 32;
        for (int v = 0; v < 8; ++v) {
          int row = mtile * 16 + v + koff;
          float pre = acc[v] + (float)xg[(xb + row) * 1024 + col];
          if (isOh) Goh[row * 1024 + col] = pre + b_o[col];
          else      Gc [row * 1024 + col] = tanhf(pre + b_c[col]);
        }
      }
    }
    ++ep; gbar(ctr, ep * NWG);

    // ================= cell update: c = f*c + i*g =================
    if (wg < 16) {
      const int base = (wg * 256 + tid) * 8;       // 4096 threads * 8 = 32768
      for (int q = 0; q < 8; ++q) {
        const int i0 = base + q;
        const float cn = Gf[i0] * cf[i0] + Gi[i0] * Gc[i0];
        cf[i0] = cn;
        cbf[i0] = (__bf16)cn;
      }
    }
    ++ep; gbar(ctr, ep * NWG);

    // ================= phase 2: o = sig(oh + c@W_oc), h = o*tanh(c) ======
    if (wg >= 48) {
      const int w = wg - 48;
      const int col = w * 128 + wave * 16 + clo;
      for (int mtile = 0; mtile < 2; ++mtile) {
        const size_t rowoff = (size_t)(mtile * 16 + clo) * 1024;
        v8f acc = {};
        for (int kt = 0; kt < 32; ++kt) {
          Frag a, b;
          const int kb = (kt << 5) + koff;
          a.u[0] = *reinterpret_cast<const u32x4*>(cbf + rowoff + kb);
          a.u[1] = *reinterpret_cast<const u32x4*>(cbf + rowoff + kb + 16);
          const __bf16* bp = lds + (size_t)(wave * 32 + kt) * 512 + lane * 16;
          b.u[0] = *reinterpret_cast<const u32x4*>(bp);
          b.u[1] = *reinterpret_cast<const u32x4*>(bp + 8);
          acc = wmma_bf16(a.v, b.v, acc);
        }
        for (int v = 0; v < 8; ++v) {
          int row = mtile * 16 + v + koff;
          float o = 1.f / (1.f + __expf(-(acc[v] + Goh[row * 1024 + col])));
          float h = o * tanhf(cf[row * 1024 + col]);
          hbf[row * 1024 + col] = (__bf16)h;
          out[row * 1024 + col] = h;               // final step's values persist
        }
      }
    }
    ++ep; gbar(ctr, ep * NWG);
  }
}

extern "C" void kernel_launch(void* const* d_in, const int* in_sizes, int n_in,
                              void* d_out, int out_size, void* d_ws, size_t ws_size,
                              hipStream_t stream) {
  const float* x   = (const float*)d_in[0];
  const float* Wix = (const float*)d_in[1];
  const float* Wih = (const float*)d_in[2];
  const float* Wic = (const float*)d_in[3];
  const float* Wfx = (const float*)d_in[4];
  const float* Wfh = (const float*)d_in[5];
  const float* Wfc = (const float*)d_in[6];
  const float* Wcx = (const float*)d_in[7];
  const float* Wch = (const float*)d_in[8];
  const float* Wox = (const float*)d_in[9];
  const float* Woh = (const float*)d_in[10];
  const float* Woc = (const float*)d_in[11];
  const float* bi  = (const float*)d_in[12];
  const float* bf  = (const float*)d_in[13];
  const float* bc  = (const float*)d_in[14];
  const float* bo  = (const float*)d_in[15];

  char* ws = (char*)d_ws;
  size_t off = 0;
  __bf16* xg  = (__bf16*)(ws + off); off += (size_t)512 * 4 * 32 * 1024 * 2; // 128MB
  __bf16* xbf = (__bf16*)(ws + off); off += (size_t)512 * 32 * 1024 * 2;     //  32MB
  __bf16* pw[11];
  for (int m = 0; m < 11; ++m) { pw[m] = (__bf16*)(ws + off); off += (size_t)1024 * 1024 * 2; }
  __bf16* hbf = (__bf16*)(ws + off); off += 65536;
  __bf16* cbf = (__bf16*)(ws + off); off += 65536;
  float* cf   = (float*)(ws + off);  off += 131072;
  float* Gi   = (float*)(ws + off);  off += 131072;
  float* Gf   = (float*)(ws + off);  off += 131072;
  float* Gc   = (float*)(ws + off);  off += 131072;
  float* Goh  = (float*)(ws + off);  off += 131072;
  unsigned* ctr = (unsigned*)(ws + off); off += 256;

  k_init<<<128, 256, 0, stream>>>(hbf, cbf, cf, ctr);
  k_cvt<<<16384, 256, 0, stream>>>(x, xbf, 512 * 32 * 1024);
  // x-side gates packed contiguously (order i,f,c,o) for k_xg:
  k_pack<<<4096, 256, 0, stream>>>(Wix, pw[0]);
  k_pack<<<4096, 256, 0, stream>>>(Wfx, pw[1]);
  k_pack<<<4096, 256, 0, stream>>>(Wcx, pw[2]);
  k_pack<<<4096, 256, 0, stream>>>(Wox, pw[3]);
  k_pack<<<4096, 256, 0, stream>>>(Wih, pw[4]);
  k_pack<<<4096, 256, 0, stream>>>(Wic, pw[5]);
  k_pack<<<4096, 256, 0, stream>>>(Wfh, pw[6]);
  k_pack<<<4096, 256, 0, stream>>>(Wfc, pw[7]);
  k_pack<<<4096, 256, 0, stream>>>(Wch, pw[8]);
  k_pack<<<4096, 256, 0, stream>>>(Woh, pw[9]);
  k_pack<<<4096, 256, 0, stream>>>(Woc, pw[10]);
  k_xg<<<32768, 256, 0, stream>>>(xbf, pw[0], xg);

  (void)hipFuncSetAttribute(reinterpret_cast<const void*>(k_lstm),
                            hipFuncAttributeMaxDynamicSharedMemorySize, 262144);
  k_lstm<<<NWG, 256, 262144, stream>>>(xg, pw[4], pw[5], pw[6], pw[7], pw[8], pw[9], pw[10],
                                       bi, bf, bc, bo, hbf, cbf, cf, Gi, Gf, Gc, Goh,
                                       ctr, (float*)d_out);
}
